// Decoder_69518340653355
// MI455X (gfx1250) — matrix operand order repaired and checked by reading, not compiled, not run on hardware
//
#include <hip/hip_runtime.h>
#include <hip/hip_bf16.h>
#include <cstdint>
#include <cstddef>

#define S_DIM 80
#define E_DIM 512
#define P_DIM 256
#define D_DIM 1024
#define A_DIM 128
#define L_DIM 32
#define K_CONV 31
#define PD_DIM 512
#define KP_DIM 5
#define B_DIM 64
#define TI_DIM 400
#define T_DIM 800

typedef __attribute__((ext_vector_type(16))) __bf16 v16bf;
typedef __attribute__((ext_vector_type(8)))  float  v8f;

__device__ __forceinline__ float sigmoidf_(float x) { return 1.f / (1.f + __expf(-x)); }

__device__ __forceinline__ float drop_mask_(unsigned idx, unsigned salt) {
  unsigned x = idx * 2654435761u + salt;
  x ^= x >> 16; x *= 2246822519u; x ^= x >> 13;
  return (x & 1u) ? 2.0f : 0.0f;
}

// CDNA5 async global->LDS copy (ASYNCcnt-tracked). Generic pointers to LDS
// carry the LDS byte offset in their low 32 bits.
__device__ __forceinline__ void async_load_b128(unsigned lds_off, const void* gaddr) {
  asm volatile("global_load_async_to_lds_b128 %0, %1, off"
               :: "v"(lds_off), "v"(gaddr) : "memory");
}
__device__ __forceinline__ void wait_asynccnt0() {
  asm volatile("s_wait_asynccnt 0x0" ::: "memory");
}

// ---------------------------------------------------------------------------
// WMMA GEMM #1:  C(64 x N) = X(64 x K) @ W^T  (+ bias[n]) ; W is (N x K) bf16
// K must be a multiple of 32. Block: 256 thr (8 waves), tile 64(M) x 64(N).
// W tile staged with async-to-LDS (byte copy), overlapped with VALU fp32->bf16
// conversion of the A tile.
// ---------------------------------------------------------------------------
__global__ __launch_bounds__(256) void gemm_xt_wmma(
    const float* __restrict__ X, const __bf16* __restrict__ W,
    const float* __restrict__ bias, float* __restrict__ C, int N, int K)
{
  __shared__ __bf16 As[64 * 32];
  __shared__ __bf16 Bs[64 * 32];
  const int tid = threadIdx.x;
  const int lane = tid & 31, w = tid >> 5;
  const int hi = lane >> 4, lm = lane & 15;
  const int n0 = blockIdx.x * 64;
  const int nf = w & 3;
  const int mf0 = (w >> 2) * 2;

  // fixed per-thread staging coordinates: row = tid>>2, 8-elem k-slot
  const int sr = tid >> 2;
  const int sk = (tid & 3) * 8;
  const float4* xsrc = reinterpret_cast<const float4*>(X + (size_t)sr * K + sk);
  uint4* adst = reinterpret_cast<uint4*>(&As[sr * 32 + sk]);
  const int gn = n0 + sr;
  const bool wvalid = gn < N;
  const __bf16* wsrc = W + (size_t)gn * K + sk;
  const unsigned bdst = (unsigned)(uintptr_t)&Bs[sr * 32 + sk];
  if (!wvalid) {
    uint4 z = {0u, 0u, 0u, 0u};
    *reinterpret_cast<uint4*>(&Bs[sr * 32 + sk]) = z;
  }

  v8f acc0 = {0.f,0.f,0.f,0.f,0.f,0.f,0.f,0.f};
  v8f acc1 = acc0;

  for (int k0 = 0; k0 < K; k0 += 32) {
    if (wvalid) async_load_b128(bdst, wsrc);   // W tile: async byte copy
    wsrc += 32;
    float4 a0 = xsrc[0], a1 = xsrc[1];         // A tile: fp32 -> bf16 pack
    xsrc += 8;
    union { __bf16 h[8]; uint4 u; } pk;
    pk.h[0] = (__bf16)a0.x; pk.h[1] = (__bf16)a0.y;
    pk.h[2] = (__bf16)a0.z; pk.h[3] = (__bf16)a0.w;
    pk.h[4] = (__bf16)a1.x; pk.h[5] = (__bf16)a1.y;
    pk.h[6] = (__bf16)a1.z; pk.h[7] = (__bf16)a1.w;
    *adst = pk.u;
    wait_asynccnt0();
    __syncthreads();

    v16bf af0, af1, bf;
    const __bf16* br  = &Bs[(nf * 16 + lm) * 32];
    const __bf16* ar0 = &As[(mf0 * 16 + lm) * 32];
    const __bf16* ar1 = &As[((mf0 + 1) * 16 + lm) * 32];
#pragma unroll
    for (int v = 0; v < 8; ++v) {
      int kb = ((v < 4) ? 2 * v : 16 + 2 * (v - 4)) + (hi ? 8 : 0); // B lane map
      bf[2 * v] = br[kb]; bf[2 * v + 1] = br[kb + 1];
      int ka = ((v < 4) ? 2 * v : 16 + 2 * (v - 4)) + (hi ? 8 : 0); // A lane map
      af0[2 * v] = ar0[ka]; af0[2 * v + 1] = ar0[ka + 1];
      af1[2 * v] = ar1[ka]; af1[2 * v + 1] = ar1[ka + 1];
    }
    acc0 = __builtin_amdgcn_wmma_f32_16x16x32_bf16(false, af0, false, bf, (short)0, acc0, false, false);
    acc1 = __builtin_amdgcn_wmma_f32_16x16x32_bf16(false, af1, false, bf, (short)0, acc1, false, false);
    __syncthreads();
  }

  int n = n0 + nf * 16 + lm;
  if (n < N) {
    float bv = bias ? bias[n] : 0.f;
#pragma unroll
    for (int v = 0; v < 8; ++v) {
      int m = mf0 * 16 + v + hi * 8;
      C[(size_t)m * N + n]        = acc0[v] + bv;
      C[(size_t)(m + 16) * N + n] = acc1[v] + bv;
    }
  }
}

// ---------------------------------------------------------------------------
// WMMA GEMM #2 (per-batch, im2col-on-the-fly conv):
//   Y[b](M x T) = W(M x Cin*KW) @ im2col(X[b]) + bias[m]
// Block 256 thr, tile 128(M) x 32(T): wave w owns m-frag w, two n-frags.
// W tile: async-to-LDS when the K chunk is full; scalar guarded tail else.
// ---------------------------------------------------------------------------
template <int KW>
__global__ __launch_bounds__(256) void conv_gemm_wmma(
    const __bf16* __restrict__ W, const float* __restrict__ X,
    const float* __restrict__ bias, float* __restrict__ Y,
    int M, int Cin, int Tlen, int pad)
{
  __shared__ __bf16 Wsm[128 * 32];
  __shared__ __bf16 Xs[32 * 32];
  const int tid = threadIdx.x, lane = tid & 31, w = tid >> 5;
  const int hi = lane >> 4, lm = lane & 15;
  const int m0 = blockIdx.x * 128, n0 = blockIdx.y * 32, b = blockIdx.z;
  const int K = Cin * KW;

  // W staging: two 16-byte chunks per thread
  const int wr = tid >> 2;
  const int wk = (tid & 3) * 8;
  const int gm0 = m0 + wr, gm1 = m0 + wr + 64;
  const bool v0 = gm0 < M, v1 = gm1 < M;
  const __bf16* wsrc0 = W + (size_t)gm0 * K + wk;
  const __bf16* wsrc1 = W + (size_t)gm1 * K + wk;
  const unsigned wdst0 = (unsigned)(uintptr_t)&Wsm[wr * 32 + wk];
  const unsigned wdst1 = (unsigned)(uintptr_t)&Wsm[(wr + 64) * 32 + wk];
  {
    uint4 z = {0u, 0u, 0u, 0u};
    if (!v0) *reinterpret_cast<uint4*>(&Wsm[wr * 32 + wk]) = z;
    if (!v1) *reinterpret_cast<uint4*>(&Wsm[(wr + 64) * 32 + wk]) = z;
  }
  // X staging coordinates (KW==1 fast path): 4 contiguous t per thread
  const int xr = tid >> 3;
  const int tcol = (tid & 7) * 4;

  v8f acc0 = {0.f,0.f,0.f,0.f,0.f,0.f,0.f,0.f};
  v8f acc1 = acc0;

  for (int k0 = 0; k0 < K; k0 += 32) {
    const bool kfull = (k0 + 32 <= K);
    if (kfull) {
      if (v0) async_load_b128(wdst0, wsrc0);
      if (v1) async_load_b128(wdst1, wsrc1);
    } else {
      for (int j = 0; j < 8; ++j) {
        int gk = k0 + wk + j;
        Wsm[wr * 32 + wk + j]        = (v0 && gk < K) ? W[(size_t)gm0 * K + gk] : (__bf16)0.f;
        Wsm[(wr + 64) * 32 + wk + j] = (v1 && gk < K) ? W[(size_t)gm1 * K + gk] : (__bf16)0.f;
      }
    }
    wsrc0 += 32; wsrc1 += 32;

    if (KW == 1) {
      int gk = k0 + xr;       // channel
      int t0 = n0 + tcol;
      if (gk < Cin && t0 + 4 <= Tlen) {
        float4 xv = *reinterpret_cast<const float4*>(X + ((size_t)b * Cin + gk) * Tlen + t0);
        union { __bf16 h[4]; uint2 u; } p;
        p.h[0] = (__bf16)xv.x; p.h[1] = (__bf16)xv.y;
        p.h[2] = (__bf16)xv.z; p.h[3] = (__bf16)xv.w;
        *reinterpret_cast<uint2*>(&Xs[xr * 32 + tcol]) = p.u;
      } else {
        for (int j = 0; j < 4; ++j) {
          int t = t0 + j;
          float val = 0.f;
          if (gk < Cin && t < Tlen) val = X[((size_t)b * Cin + gk) * Tlen + t];
          Xs[xr * 32 + tcol + j] = (__bf16)val;
        }
      }
    } else {
      for (int j = 0; j < 4; ++j) {
        int i = tid * 4 + j;
        int k = i >> 5, n = i & 31;
        int gk = k0 + k;
        float val = 0.f;
        if (gk < K) {
          int ic = gk / KW, r = gk - ic * KW;
          int t = n0 + n + r - pad;
          if (t >= 0 && t < Tlen) val = X[((size_t)b * Cin + ic) * Tlen + t];
        }
        Xs[i] = (__bf16)val;
      }
    }
    if (kfull) wait_asynccnt0();
    __syncthreads();

    v16bf af, bf0, bf1;
    const __bf16* ar = &Wsm[(w * 16 + lm) * 32];
#pragma unroll
    for (int v = 0; v < 8; ++v) {
      int ka = ((v < 4) ? 2 * v : 16 + 2 * (v - 4)) + (hi ? 8 : 0);
      af[2 * v] = ar[ka]; af[2 * v + 1] = ar[ka + 1];
      int kb = ((v < 4) ? 2 * v : 16 + 2 * (v - 4)) + (hi ? 8 : 0);
      bf0[2 * v]     = Xs[kb * 32 + lm];
      bf0[2 * v + 1] = Xs[(kb + 1) * 32 + lm];
      bf1[2 * v]     = Xs[kb * 32 + 16 + lm];
      bf1[2 * v + 1] = Xs[(kb + 1) * 32 + 16 + lm];
    }
    acc0 = __builtin_amdgcn_wmma_f32_16x16x32_bf16(false, af, false, bf0, (short)0, acc0, false, false);
    acc1 = __builtin_amdgcn_wmma_f32_16x16x32_bf16(false, af, false, bf1, (short)0, acc1, false, false);
    __syncthreads();
  }

#pragma unroll
  for (int v = 0; v < 8; ++v) {
    int m = m0 + w * 16 + v + hi * 8;
    if (m < M) {
      float bv = bias ? bias[m] : 0.f;
      int t0 = n0 + lm, t1 = n0 + 16 + lm;
      if (t0 < Tlen) Y[((size_t)b * M + m) * Tlen + t0] = acc0[v] + bv;
      if (t1 < Tlen) Y[((size_t)b * M + m) * Tlen + t1] = acc1[v] + bv;
    }
  }
}

// ------------------------- prep / elementwise kernels ----------------------
__global__ void cvt_bf16_k(const float* __restrict__ s, __bf16* __restrict__ d, size_t n) {
  size_t i = (size_t)blockIdx.x * blockDim.x + threadIdx.x;
  if (i < n) d[i] = (__bf16)s[i];
}
__global__ void cat_rows_bf16_k(const float* __restrict__ A, int k1,
                                const float* __restrict__ B, int k2,
                                __bf16* __restrict__ d, int N) {
  size_t i = (size_t)blockIdx.x * blockDim.x + threadIdx.x;
  int K = k1 + k2;
  if (i >= (size_t)N * K) return;
  int n = (int)(i / K), k = (int)(i % K);
  d[i] = (__bf16)((k < k1) ? A[(size_t)n * k1 + k] : B[(size_t)n * k2 + (k - k1)]);
}
__global__ void add_vec_k(const float* a, const float* b, float* o, int n) {
  int i = blockIdx.x * blockDim.x + threadIdx.x;
  if (i < n) o[i] = a[i] + b[i];
}
__global__ void cat_fs_w_k(const float* fw, const float* sw, __bf16* d) {
  size_t i = (size_t)blockIdx.x * blockDim.x + threadIdx.x;
  const int K = D_DIM + E_DIM;
  if (i >= (size_t)(S_DIM + 1) * K) return;
  int n = (int)(i / K), k = (int)(i % K);
  d[i] = (__bf16)((n < S_DIM) ? fw[(size_t)n * K + k] : sw[k]);
}
__global__ void cat_fs_b_k(const float* fb, const float* sb, float* d) {
  int i = blockIdx.x * blockDim.x + threadIdx.x;
  if (i < S_DIM) d[i] = fb[i]; else if (i == S_DIM) d[i] = sb[0];
}
__global__ void zero_f_k(float* p, size_t n) {
  size_t i = (size_t)blockIdx.x * blockDim.x + threadIdx.x;
  if (i < n) p[i] = 0.f;
}
__global__ void relu_drop_k(float* x, size_t n, unsigned salt) {
  size_t i = (size_t)blockIdx.x * blockDim.x + threadIdx.x;
  if (i >= n) return;
  float v = x[i] * drop_mask_((unsigned)i, salt);
  x[i] = v > 0.f ? v : 0.f;
}

// ------------------------- decoder step kernels ----------------------------
__global__ void build_xcat1_k(const float* __restrict__ pre, const float* __restrict__ ctx,
                              const float* __restrict__ h1, float* __restrict__ xc, int step) {
  int i = blockIdx.x * blockDim.x + threadIdx.x;
  const int K = P_DIM + E_DIM + D_DIM;
  if (i >= B_DIM * K) return;
  int b = i / K, j = i % K;
  float v;
  if (j < P_DIM)              v = pre[((size_t)b * P_DIM + j) * T_DIM + step];
  else if (j < P_DIM + E_DIM) v = ctx[b * E_DIM + (j - P_DIM)];
  else                        v = h1[b * D_DIM + (j - P_DIM - E_DIM)];
  xc[i] = v;
}
__global__ void concat2_k(const float* A, int n1, const float* B, int n2, float* o) {
  int i = blockIdx.x * blockDim.x + threadIdx.x;
  int n = n1 + n2;
  if (i >= B_DIM * n) return;
  int b = i / n, j = i % n;
  o[i] = (j < n1) ? A[(size_t)b * n1 + j] : B[(size_t)b * n2 + (j - n1)];
}
__global__ void lstm_cell_k(const float* __restrict__ g, float* __restrict__ h, float* __restrict__ c) {
  int i = blockIdx.x * blockDim.x + threadIdx.x;
  if (i >= B_DIM * D_DIM) return;
  int b = i / D_DIM, j = i % D_DIM;
  const float* gr = g + (size_t)b * 4 * D_DIM;
  float gi = gr[j], gf = gr[D_DIM + j], gg = gr[2 * D_DIM + j], go = gr[3 * D_DIM + j];
  float cc = sigmoidf_(gf) * c[i] + sigmoidf_(gi) * tanhf(gg);
  c[i] = cc;
  h[i] = sigmoidf_(go) * tanhf(cc);
}
__global__ void att_energy_k(const float* __restrict__ cum, const float* __restrict__ Vh,
                             const float* __restrict__ Ws, const float* __restrict__ Uw,
                             const float* __restrict__ Fw, const float* __restrict__ vw,
                             const unsigned char* __restrict__ emask, float* __restrict__ e) {
  int b = blockIdx.y;
  int t = blockIdx.x * 64 + threadIdx.x;
  if (t >= TI_DIM) return;
  const float* cb = cum + (size_t)b * TI_DIM;
  float loc[L_DIM];
  for (int l = 0; l < L_DIM; ++l) {
    float s = 0.f;
    for (int k = 0; k < K_CONV; ++k) {
      int tt = t + k - K_CONV / 2;
      if (tt >= 0 && tt < TI_DIM) s += Fw[l * K_CONV + k] * cb[tt];
    }
    loc[l] = s;
  }
  const float* wsb = Ws + (size_t)b * A_DIM;
  const float* vhb = Vh + (size_t)b * A_DIM * TI_DIM;
  float en = 0.f;
  for (int a = 0; a < A_DIM; ++a) {
    const float* ur = Uw + a * L_DIM;
    float s = 0.f;
    for (int l = 0; l < L_DIM; ++l) s += ur[l] * loc[l];
    en += vw[a] * tanhf(wsb[a] + vhb[(size_t)a * TI_DIM + t] + s);
  }
  e[(size_t)b * TI_DIM + t] = emask[(size_t)b * TI_DIM + t] ? -INFINITY : en;
}
__global__ void softmax_k(const float* __restrict__ e, float* __restrict__ aw,
                          float* __restrict__ cum, float* __restrict__ attw, int step) {
  __shared__ float red[256];
  int b = blockIdx.x, tid = threadIdx.x;
  const float* eb = e + (size_t)b * TI_DIM;
  float mx = -INFINITY;
  for (int t = tid; t < TI_DIM; t += 256) mx = fmaxf(mx, eb[t]);
  red[tid] = mx; __syncthreads();
  for (int s = 128; s > 0; s >>= 1) { if (tid < s) red[tid] = fmaxf(red[tid], red[tid + s]); __syncthreads(); }
  mx = red[0]; __syncthreads();
  float sum = 0.f;
  for (int t = tid; t < TI_DIM; t += 256) sum += __expf(eb[t] - mx);
  red[tid] = sum; __syncthreads();
  for (int s = 128; s > 0; s >>= 1) { if (tid < s) red[tid] += red[tid + s]; __syncthreads(); }
  float inv = 1.f / red[0];
  for (int t = tid; t < TI_DIM; t += 256) {
    float a = __expf(eb[t] - mx) * inv;
    aw[(size_t)b * TI_DIM + t] = a;
    cum[(size_t)b * TI_DIM + t] += a;
    attw[((size_t)b * TI_DIM + t) * T_DIM + step] = a;
  }
}
__global__ void ctx_k(const float* __restrict__ enc, const float* __restrict__ aw,
                      float* __restrict__ ctx) {
  int i = blockIdx.x * blockDim.x + threadIdx.x;
  if (i >= B_DIM * E_DIM) return;
  int b = i / E_DIM, e = i % E_DIM;
  const float* er = enc + ((size_t)b * E_DIM + e) * TI_DIM;
  const float* ar = aw + (size_t)b * TI_DIM;
  float s = 0.f;
  for (int t = 0; t < TI_DIM; ++t) s += er[t] * ar[t];
  ctx[i] = s;
}
__global__ void featstop_store_k(const float* __restrict__ fs, float* __restrict__ feat,
                                 float* __restrict__ stop, int step) {
  int i = blockIdx.x * blockDim.x + threadIdx.x;
  if (i >= B_DIM * (S_DIM + 1)) return;
  int b = i / (S_DIM + 1), s = i % (S_DIM + 1);
  float v = fs[(size_t)b * (S_DIM + 1) + s];
  if (s < S_DIM) feat[((size_t)b * S_DIM + s) * T_DIM + step] = tanhf(v);
  else           stop[(size_t)b * T_DIM + step] = v;
}

// ------------------------- postnet batchnorm -------------------------------
__global__ void bn_stats_k(const float* __restrict__ y, float* __restrict__ st) {
  __shared__ float s1[256], s2[256];
  int c = blockIdx.x, tid = threadIdx.x;
  float sum = 0.f, sq = 0.f;
  for (int i = tid; i < B_DIM * T_DIM; i += 256) {
    int b = i / T_DIM, t = i % T_DIM;
    float v = y[((size_t)b * PD_DIM + c) * T_DIM + t];
    sum += v; sq += v * v;
  }
  s1[tid] = sum; s2[tid] = sq; __syncthreads();
  for (int s = 128; s > 0; s >>= 1) {
    if (tid < s) { s1[tid] += s1[tid + s]; s2[tid] += s2[tid + s]; }
    __syncthreads();
  }
  if (tid == 0) {
    float m = s1[0] / (float)(B_DIM * T_DIM);
    float var = s2[0] / (float)(B_DIM * T_DIM) - m * m;
    st[c] = m;
    st[PD_DIM + c] = rsqrtf(var + 1e-5f);
  }
}
__global__ void bn_apply_tanh_k(float* __restrict__ y, const float* __restrict__ st,
                                const float* __restrict__ g, const float* __restrict__ b) {
  size_t i = (size_t)blockIdx.x * blockDim.x + threadIdx.x;
  const size_t n = (size_t)B_DIM * PD_DIM * T_DIM;
  if (i >= n) return;
  int c = (int)((i / T_DIM) % PD_DIM);
  y[i] = tanhf(g[c] * (y[i] - st[c]) * st[PD_DIM + c] + b[c]);
}

// ------------------------- finalize masking --------------------------------
__global__ void finalize_mask_k(float* __restrict__ out, const unsigned char* __restrict__ dm) {
  size_t i = (size_t)blockIdx.x * blockDim.x + threadIdx.x;
  const size_t FEATN = (size_t)B_DIM * S_DIM * T_DIM;
  const size_t STOPO = 2 * FEATN;
  const size_t ATTO  = STOPO + (size_t)B_DIM * T_DIM;
  const size_t TOT   = ATTO + (size_t)B_DIM * TI_DIM * T_DIM;
  if (i >= TOT) return;
  if (i < FEATN) {
    int t = (int)(i % T_DIM); int b = (int)(i / ((size_t)S_DIM * T_DIM));
    if (dm[(size_t)b * T_DIM + t]) out[i] = -1.0f;
  } else if (i < STOPO) {
    size_t r = i - FEATN;
    int t = (int)(r % T_DIM); int b = (int)(r / ((size_t)S_DIM * T_DIM));
    if (dm[(size_t)b * T_DIM + t]) out[i] = 0.f;
  } else if (i < ATTO) {
    size_t r = i - STOPO;
    if (dm[r]) out[i] = 1000.f;
  } else {
    size_t r = i - ATTO;
    int t = (int)(r % T_DIM); int b = (int)(r / ((size_t)TI_DIM * T_DIM));
    if (dm[(size_t)b * T_DIM + t]) out[i] = 0.f;
  }
}

// ===========================================================================
extern "C" void kernel_launch(void* const* d_in, const int* in_sizes, int n_in,
                              void* d_out, int out_size, void* d_ws, size_t ws_size,
                              hipStream_t stream) {
  (void)in_sizes; (void)n_in; (void)out_size; (void)ws_size;
  const float* enc    = (const float*)d_in[0];
  const unsigned char* emask = (const unsigned char*)d_in[1];
  const float* spec   = (const float*)d_in[2];
  const unsigned char* dmask = (const unsigned char*)d_in[3];
  const float* pre_w1 = (const float*)d_in[4];
  const float* pre_b1 = (const float*)d_in[5];
  const float* pre_w2 = (const float*)d_in[6];
  const float* pre_b2 = (const float*)d_in[7];
  const float* att_Ww = (const float*)d_in[8];
  const float* att_Wb = (const float*)d_in[9];
  const float* att_Vw = (const float*)d_in[10];
  const float* att_Uw = (const float*)d_in[11];
  const float* att_Fw = (const float*)d_in[12];
  const float* att_vw = (const float*)d_in[13];
  const float* w_ih1  = (const float*)d_in[14];
  const float* w_hh1  = (const float*)d_in[15];
  const float* b_ih1  = (const float*)d_in[16];
  const float* b_hh1  = (const float*)d_in[17];
  const float* w_ih2  = (const float*)d_in[18];
  const float* w_hh2  = (const float*)d_in[19];
  const float* b_ih2  = (const float*)d_in[20];
  const float* b_hh2  = (const float*)d_in[21];
  const float* feat_w = (const float*)d_in[22];
  const float* feat_b = (const float*)d_in[23];
  const float* stop_w = (const float*)d_in[24];
  const float* stop_b = (const float*)d_in[25];
  const float* post_in_w  = (const float*)d_in[26];
  const float* post_in_b  = (const float*)d_in[27];
  const float* bn_in_g    = (const float*)d_in[28];
  const float* bn_in_b    = (const float*)d_in[29];
  const float* post_body_w = (const float*)d_in[30];
  const float* post_body_b = (const float*)d_in[31];
  const float* bn_body_g   = (const float*)d_in[32];
  const float* bn_body_b   = (const float*)d_in[33];
  const float* post_out_w  = (const float*)d_in[34];
  const float* post_out_b  = (const float*)d_in[35];

  // d_out segments: feat | residual | stop | attw
  float* out       = (float*)d_out;
  const size_t FEATN = (size_t)B_DIM * S_DIM * T_DIM;
  float* feat_out  = out;
  float* resid_out = out + FEATN;
  float* stop_out  = out + 2 * FEATN;
  float* attw_out  = out + 2 * FEATN + (size_t)B_DIM * T_DIM;

  // ---- workspace allocator ----
  char* base = (char*)d_ws;
  size_t off = 0;
  auto alloc = [&](size_t bytes) -> void* {
    off = (off + 255) & ~(size_t)255;
    void* p = base + off;
    off += bytes;
    return p;
  };
  const int K1 = P_DIM + E_DIM + D_DIM;   // 1792
  const int K2 = 2 * D_DIM;               // 2048
  const int KFS = D_DIM + E_DIM;          // 1536
  __bf16* wcat1   = (__bf16*)alloc((size_t)4 * D_DIM * K1 * 2);
  __bf16* wcat2   = (__bf16*)alloc((size_t)4 * D_DIM * K2 * 2);
  __bf16* wWs     = (__bf16*)alloc((size_t)A_DIM * D_DIM * 2);
  __bf16* wfs     = (__bf16*)alloc((size_t)(S_DIM + 1) * KFS * 2);
  __bf16* wpre1   = (__bf16*)alloc((size_t)P_DIM * S_DIM * 2);
  __bf16* wpre2   = (__bf16*)alloc((size_t)P_DIM * P_DIM * 2);
  __bf16* wVa     = (__bf16*)alloc((size_t)A_DIM * E_DIM * 2);
  __bf16* wpostin = (__bf16*)alloc((size_t)PD_DIM * S_DIM * 2);
  __bf16* wpbody  = (__bf16*)alloc((size_t)PD_DIM * PD_DIM * KP_DIM * 2);
  __bf16* wpout   = (__bf16*)alloc((size_t)S_DIM * PD_DIM * 2);
  float* bias1  = (float*)alloc(4 * D_DIM * 4);
  float* bias2  = (float*)alloc(4 * D_DIM * 4);
  float* biasfs = (float*)alloc((S_DIM + 1) * 4);
  // big ping-pong buffers (postnet); prenet temporaries alias them
  float* postA = (float*)alloc((size_t)B_DIM * PD_DIM * T_DIM * 4);
  float* postB = (float*)alloc((size_t)B_DIM * PD_DIM * T_DIM * 4);
  float* preA  = postA;                               // (B,P,T) dead before postnet
  float* preB  = postB;                               // prenet output, dead before postnet
  float* Vh    = (float*)alloc((size_t)B_DIM * A_DIM * TI_DIM * 4);
  // states (zeroed each call, contiguous)
  const size_t STATEN = (size_t)B_DIM * (4 * D_DIM + E_DIM + TI_DIM);
  float* h1  = (float*)alloc(STATEN * 4);
  float* c1  = h1 + (size_t)B_DIM * D_DIM;
  float* h2  = c1 + (size_t)B_DIM * D_DIM;
  float* c2  = h2 + (size_t)B_DIM * D_DIM;
  float* ctx = c2 + (size_t)B_DIM * D_DIM;
  float* cum = ctx + (size_t)B_DIM * E_DIM;
  float* gates1 = (float*)alloc((size_t)B_DIM * 4 * D_DIM * 4);
  float* gates2 = (float*)alloc((size_t)B_DIM * 4 * D_DIM * 4);
  float* xcat1  = (float*)alloc((size_t)B_DIM * K1 * 4);
  float* xcat2  = (float*)alloc((size_t)B_DIM * K2 * 4);
  float* lin    = (float*)alloc((size_t)B_DIM * KFS * 4);
  float* Wsout  = (float*)alloc((size_t)B_DIM * A_DIM * 4);
  float* fsout  = (float*)alloc((size_t)B_DIM * (S_DIM + 1) * 4);
  float* ebuf   = (float*)alloc((size_t)B_DIM * TI_DIM * 4);
  float* awbuf  = (float*)alloc((size_t)B_DIM * TI_DIM * 4);
  float* bnst   = (float*)alloc(2 * PD_DIM * 4);

  auto g1 = [](size_t n) { return dim3((unsigned)((n + 255) / 256)); };

  // ---- weight prep (bf16, fused [w_ih | w_hh]) ----
  cat_rows_bf16_k<<<g1((size_t)4 * D_DIM * K1), 256, 0, stream>>>(w_ih1, P_DIM + E_DIM, w_hh1, D_DIM, wcat1, 4 * D_DIM);
  cat_rows_bf16_k<<<g1((size_t)4 * D_DIM * K2), 256, 0, stream>>>(w_ih2, D_DIM, w_hh2, D_DIM, wcat2, 4 * D_DIM);
  cvt_bf16_k<<<g1((size_t)A_DIM * D_DIM), 256, 0, stream>>>(att_Ww, wWs, (size_t)A_DIM * D_DIM);
  cat_fs_w_k<<<g1((size_t)(S_DIM + 1) * KFS), 256, 0, stream>>>(feat_w, stop_w, wfs);
  cvt_bf16_k<<<g1((size_t)P_DIM * S_DIM), 256, 0, stream>>>(pre_w1, wpre1, (size_t)P_DIM * S_DIM);
  cvt_bf16_k<<<g1((size_t)P_DIM * P_DIM), 256, 0, stream>>>(pre_w2, wpre2, (size_t)P_DIM * P_DIM);
  cvt_bf16_k<<<g1((size_t)A_DIM * E_DIM), 256, 0, stream>>>(att_Vw, wVa, (size_t)A_DIM * E_DIM);
  cvt_bf16_k<<<g1((size_t)PD_DIM * S_DIM), 256, 0, stream>>>(post_in_w, wpostin, (size_t)PD_DIM * S_DIM);
  cvt_bf16_k<<<g1((size_t)PD_DIM * PD_DIM * KP_DIM), 256, 0, stream>>>(post_body_w, wpbody, (size_t)PD_DIM * PD_DIM * KP_DIM);
  cvt_bf16_k<<<g1((size_t)S_DIM * PD_DIM), 256, 0, stream>>>(post_out_w, wpout, (size_t)S_DIM * PD_DIM);
  add_vec_k<<<g1(4 * D_DIM), 256, 0, stream>>>(b_ih1, b_hh1, bias1, 4 * D_DIM);
  add_vec_k<<<g1(4 * D_DIM), 256, 0, stream>>>(b_ih2, b_hh2, bias2, 4 * D_DIM);
  cat_fs_b_k<<<1, 128, 0, stream>>>(feat_b, stop_b, biasfs);
  zero_f_k<<<g1(STATEN), 256, 0, stream>>>(h1, STATEN);

  // ---- prenet: (B,P,T) via per-batch WMMA GEMM ----
  conv_gemm_wmma<1><<<dim3(2, (T_DIM + 31) / 32, B_DIM), 256, 0, stream>>>(
      wpre1, spec, pre_b1, preA, P_DIM, S_DIM, T_DIM, 0);
  relu_drop_k<<<g1((size_t)B_DIM * P_DIM * T_DIM), 256, 0, stream>>>(preA, (size_t)B_DIM * P_DIM * T_DIM, 0x9E3779B9u);
  conv_gemm_wmma<1><<<dim3(2, (T_DIM + 31) / 32, B_DIM), 256, 0, stream>>>(
      wpre2, preA, pre_b2, preB, P_DIM, P_DIM, T_DIM, 0);
  relu_drop_k<<<g1((size_t)B_DIM * P_DIM * T_DIM), 256, 0, stream>>>(preB, (size_t)B_DIM * P_DIM * T_DIM, 0x85EBCA6Bu);

  // ---- Vh = att_Vw @ encoder_feature : (B,A,TI) ----
  conv_gemm_wmma<1><<<dim3(1, (TI_DIM + 31) / 32, B_DIM), 256, 0, stream>>>(
      wVa, enc, nullptr, Vh, A_DIM, E_DIM, TI_DIM, 0);

  // ---- sequential decoder scan (weights stay L2-resident across 800 steps) ----
  for (int t = 0; t < T_DIM; ++t) {
    build_xcat1_k<<<g1((size_t)B_DIM * K1), 256, 0, stream>>>(preB, ctx, h1, xcat1, t);
    gemm_xt_wmma<<<dim3(4 * D_DIM / 64), 256, 0, stream>>>(xcat1, wcat1, bias1, gates1, 4 * D_DIM, K1);
    lstm_cell_k<<<g1((size_t)B_DIM * D_DIM), 256, 0, stream>>>(gates1, h1, c1);
    concat2_k<<<g1((size_t)B_DIM * K2), 256, 0, stream>>>(h1, D_DIM, h2, D_DIM, xcat2);
    gemm_xt_wmma<<<dim3(4 * D_DIM / 64), 256, 0, stream>>>(xcat2, wcat2, bias2, gates2, 4 * D_DIM, K2);
    lstm_cell_k<<<g1((size_t)B_DIM * D_DIM), 256, 0, stream>>>(gates2, h2, c2);
    gemm_xt_wmma<<<dim3(A_DIM / 64), 256, 0, stream>>>(h2, wWs, att_Wb, Wsout, A_DIM, D_DIM);
    att_energy_k<<<dim3((TI_DIM + 63) / 64, B_DIM), 64, 0, stream>>>(cum, Vh, Wsout, att_Uw, att_Fw, att_vw, emask, ebuf);
    softmax_k<<<dim3(B_DIM), 256, 0, stream>>>(ebuf, awbuf, cum, attw_out, t);
    ctx_k<<<g1((size_t)B_DIM * E_DIM), 256, 0, stream>>>(enc, awbuf, ctx);
    concat2_k<<<g1((size_t)B_DIM * KFS), 256, 0, stream>>>(h2, D_DIM, ctx, E_DIM, lin);
    gemm_xt_wmma<<<dim3((S_DIM + 1 + 63) / 64), 256, 0, stream>>>(lin, wfs, biasfs, fsout, S_DIM + 1, KFS);
    featstop_store_k<<<g1((size_t)B_DIM * (S_DIM + 1)), 256, 0, stream>>>(fsout, feat_out, stop_out, t);
  }

  // ---- postnet ----
  conv_gemm_wmma<1><<<dim3(PD_DIM / 128, T_DIM / 32, B_DIM), 256, 0, stream>>>(
      wpostin, feat_out, post_in_b, postA, PD_DIM, S_DIM, T_DIM, 0);
  bn_stats_k<<<dim3(PD_DIM), 256, 0, stream>>>(postA, bnst);
  bn_apply_tanh_k<<<g1((size_t)B_DIM * PD_DIM * T_DIM), 256, 0, stream>>>(postA, bnst, bn_in_g, bn_in_b);
  float* ycur = postA;
  float* ynext = postB;
  for (int l = 0; l < 5; ++l) {
    conv_gemm_wmma<KP_DIM><<<dim3(PD_DIM / 128, T_DIM / 32, B_DIM), 256, 0, stream>>>(
        wpbody, ycur, post_body_b, ynext, PD_DIM, PD_DIM, T_DIM, KP_DIM / 2);
    bn_stats_k<<<dim3(PD_DIM), 256, 0, stream>>>(ynext, bnst);
    bn_apply_tanh_k<<<g1((size_t)B_DIM * PD_DIM * T_DIM), 256, 0, stream>>>(ynext, bnst, bn_body_g, bn_body_b);
    float* tmp = ycur; ycur = ynext; ynext = tmp;
  }
  conv_gemm_wmma<1><<<dim3(1, T_DIM / 32, B_DIM), 256, 0, stream>>>(
      wpout, ycur, post_out_b, resid_out, S_DIM, PD_DIM, T_DIM, 0);

  // ---- apply decoder mask to all outputs ----
  const size_t TOT = 2 * FEATN + (size_t)B_DIM * T_DIM + (size_t)B_DIM * TI_DIM * T_DIM;
  finalize_mask_k<<<g1(TOT), 256, 0, stream>>>(out, dmask);
}
